// RWKVTimeMixingChunked_6803228197623
// MI455X (gfx1250) — compile-verified, matrix operand order checked by reading, not run on hardware
//
#include <hip/hip_runtime.h>
#include <hip/hip_bf16.h>
#include <cstdint>
#include <cstddef>

// RWKV time-mixing for MI455X (gfx1250, wave32).
// Big GEMM runs on v_wmma_f32_16x16x32_bf16 with double-buffered LDS panels
// filled by CDNA5 async global->LDS copies (ASYNCcnt); elementwise math fused
// into GEMM epilogues; T-sequential recurrence as a tiny scalar kernel.

typedef __attribute__((ext_vector_type(16))) __bf16 v16bf;
typedef __attribute__((ext_vector_type(8)))  float  v8f;
typedef __attribute__((ext_vector_type(4)))  int    v4i;

#define B_     8
#define T_     4096
#define D_     2048
#define S_     64
#define SHIFT_ 2048            // int(T*0.5)
#define LN_EPS 1e-5f

// ---------------------------------------------------------------------------
// CDNA5 async global->LDS copy helpers (per-lane 16B, tracked by ASYNCcnt)
// Builtin prototype (from hipcc diagnostic): (v4i AS1*, v4i AS3*, imm, imm)
// ---------------------------------------------------------------------------
__device__ __forceinline__ void async_copy16(const __bf16* g, __bf16* l) {
#if __has_builtin(__builtin_amdgcn_global_load_async_to_lds_b128)
  __builtin_amdgcn_global_load_async_to_lds_b128(
      (__attribute__((address_space(1))) v4i*)g,
      (__attribute__((address_space(3))) v4i*)l, 0, 0);
#else
  *(uint4*)l = *(const uint4*)g;   // synchronous fallback
#endif
}

__device__ __forceinline__ void async_wait_all() {
#if __has_builtin(__builtin_amdgcn_s_wait_asynccnt)
  __builtin_amdgcn_s_wait_asynccnt(0);
#else
  asm volatile("s_wait_asynccnt 0x0" ::: "memory");
#endif
}

// ---------------------------------------------------------------------------
// K0: f32 -> bf16 conversion (vectorized)
// ---------------------------------------------------------------------------
__global__ void cvt_f32_bf16(const float* __restrict__ in,
                             __bf16* __restrict__ out, int n) {
  int i = (blockIdx.x * blockDim.x + threadIdx.x) * 4;
  if (i + 3 < n) {
    float4 f = *(const float4*)(in + i);
    out[i + 0] = (__bf16)f.x;
    out[i + 1] = (__bf16)f.y;
    out[i + 2] = (__bf16)f.z;
    out[i + 3] = (__bf16)f.w;
  } else {
    for (; i < n; ++i) out[i] = (__bf16)in[i];
  }
}

// ---------------------------------------------------------------------------
// K1: x_shift = (roll(x,-SHIFT) @ Wts^T) * sigmoid(gate) + x * (1-sigmoid(gate))
//
// LDS-tiled double-buffered WMMA GEMM:
//   WG tile 64(M) x 128(N), K-stage 64, 256 threads = 8 waves.
//   Wave (wm, wn) with wm in [0,2), wn in [0,4) owns a 2x2 grid of 16x16 tiles.
//   Panels staged with async global->LDS copies; WMMA reads via ds_load.
// ---------------------------------------------------------------------------
#define BM 64
#define BN 128
#define BK 64
#define NKSTAGES (D_ / BK)   // 32

__global__ __launch_bounds__(256, 1)
void gemm_shift_gate(const __bf16* __restrict__ xb,     // [B*T, D] bf16
                     const __bf16* __restrict__ Wts_b,  // [D, D]  row = out dim e
                     const float*  __restrict__ x,      // [B*T, D] f32 original
                     const float*  __restrict__ gate,   // [D]
                     __bf16*       __restrict__ xshift) // [B*T, D] bf16 out
{
  __shared__ __bf16 As[2][BM][BK];   // 16 KB
  __shared__ __bf16 Bs[2][BN][BK];   // 32 KB

  const int rowbase = blockIdx.y * BM;   // 512 row tiles
  const int colbase = blockIdx.x * BN;   // 16 col tiles
  const int tid  = threadIdx.x;
  const int wave = tid >> 5;
  const int lane = tid & 31;
  const int lm   = lane & 15;
  const int lh   = lane >> 4;
  const int wm   = wave >> 2;            // 0..1
  const int wn   = wave & 3;             // 0..3

  const v8f vzero = {0.f, 0.f, 0.f, 0.f, 0.f, 0.f, 0.f, 0.f};
  v8f acc[2][2];
#pragma unroll
  for (int i = 0; i < 2; ++i)
#pragma unroll
    for (int j = 0; j < 2; ++j) acc[i][j] = vzero;

  // ---- panel staging (async, per-lane 16B chunks) ----
  auto stage = [&](int buf, int k0) {
    // A panel: 64 rows x 64 k = 8 KB -> 512 chunks, 2 per thread
#pragma unroll
    for (int it = 0; it < 2; ++it) {
      const int c  = tid + it * 256;
      const int m  = c >> 3;
      const int kc = (c & 7) * 8;
      const int grow = rowbase + m;
      const int bb = grow / T_;
      const int tt = grow - bb * T_;
      const int trow = bb * T_ + ((tt + SHIFT_) & (T_ - 1));  // token roll
      async_copy16(xb + (size_t)trow * D_ + k0 + kc, &As[buf][m][kc]);
    }
    // B panel: 128 rows x 64 k = 16 KB -> 1024 chunks, 4 per thread
#pragma unroll
    for (int it = 0; it < 4; ++it) {
      const int c  = tid + it * 256;
      const int n  = c >> 3;
      const int kc = (c & 7) * 8;
      async_copy16(Wts_b + (size_t)(colbase + n) * D_ + k0 + kc,
                   &Bs[buf][n][kc]);
    }
  };

  auto compute = [&](int buf) {
#pragma unroll
    for (int kk = 0; kk < BK; kk += 32) {
      v16bf afr[2], bfr[2];
#pragma unroll
      for (int i = 0; i < 2; ++i)
        afr[i] = *(const v16bf*)&As[buf][wm * 32 + i * 16 + lm][kk + lh * 16];
#pragma unroll
      for (int j = 0; j < 2; ++j)
        bfr[j] = *(const v16bf*)&Bs[buf][wn * 32 + j * 16 + lm][kk + lh * 16];
#pragma unroll
      for (int i = 0; i < 2; ++i)
#pragma unroll
        for (int j = 0; j < 2; ++j)
          acc[i][j] = __builtin_amdgcn_wmma_f32_16x16x32_bf16(
              false, afr[i], false, bfr[j], (short)0, acc[i][j], false, false);
    }
  };

  // ---- double-buffered main loop ----
  stage(0, 0);
  async_wait_all();
  __syncthreads();

  for (int ks = 0; ks < NKSTAGES; ++ks) {
    const int cur = ks & 1;
    if (ks + 1 < NKSTAGES) stage(cur ^ 1, (ks + 1) * BK);
    compute(cur);
    if (ks + 1 < NKSTAGES) {
      async_wait_all();   // next buffer resident
      __syncthreads();    // all waves done reading current buffer
    }
  }

  // ---- epilogue: sigmoid-gate blend with original x, store bf16 ----
#pragma unroll
  for (int j = 0; j < 2; ++j) {
    const int e = colbase + wn * 32 + j * 16 + lm;   // C layout: N = lane&15
    const float gs = 1.f / (1.f + __expf(-gate[e]));
#pragma unroll
    for (int i = 0; i < 2; ++i) {
#pragma unroll
      for (int r = 0; r < 8; ++r) {
        const int grow = rowbase + wm * 32 + i * 16 + lh * 8 + r;  // M = r+8*half
        const float xo = x[(size_t)grow * D_ + e];
        xshift[(size_t)grow * D_ + e] = (__bf16)(acc[i][j][r] * gs + xo * (1.f - gs));
      }
    }
  }
}

// ---------------------------------------------------------------------------
// K2: row-wise LayerNorm, in place on the bf16 buffer (one WG per row)
// ---------------------------------------------------------------------------
__global__ __launch_bounds__(256, 1)
void layernorm_inplace(__bf16* __restrict__ xs,          // [B*T, D] in/out
                       const float* __restrict__ ln_w,
                       const float* __restrict__ ln_b)
{
  __shared__ float red[256];
  const int tid = threadIdx.x;
  __bf16* p = xs + (size_t)blockIdx.x * D_;

  float v[8];
  float s = 0.f;
#pragma unroll
  for (int i = 0; i < 8; ++i) { v[i] = (float)p[tid + i * 256]; s += v[i]; }
  red[tid] = s; __syncthreads();
  for (int off = 128; off > 0; off >>= 1) {
    if (tid < off) red[tid] += red[tid + off];
    __syncthreads();
  }
  const float mu = red[0] * (1.f / D_);
  __syncthreads();

  float s2 = 0.f;
#pragma unroll
  for (int i = 0; i < 8; ++i) { const float d = v[i] - mu; s2 += d * d; }
  red[tid] = s2; __syncthreads();
  for (int off = 128; off > 0; off >>= 1) {
    if (tid < off) red[tid] += red[tid + off];
    __syncthreads();
  }
  const float rstd = __frsqrt_rn(red[0] * (1.f / D_) + LN_EPS);

#pragma unroll
  for (int i = 0; i < 8; ++i) {
    const int e = tid + i * 256;
    p[e] = (__bf16)((v[i] - mu) * rstd * ln_w[e] + ln_b[e]);
  }
}

// ---------------------------------------------------------------------------
// K3: k = xn@Wk^T, v = xn@Wv^T (WMMA, register software-pipelined),
//     fused wkv = exp(-exp(tf)*sigmoid(k))*v
// ---------------------------------------------------------------------------
__global__ __launch_bounds__(128, 1)
void kv_wkv(const __bf16* __restrict__ xn,    // [B*T, D]
            const __bf16* __restrict__ Wk_b,  // [S, D]
            const __bf16* __restrict__ Wv_b,  // [S, D]
            const float*  __restrict__ time_first, // [S]
            float*        __restrict__ wkv)   // [B*T, S]
{
  const int rt   = blockIdx.x;
  const int wave = threadIdx.x >> 5;     // 0..3 -> S col tile
  const int lane = threadIdx.x & 31;
  const int lm   = lane & 15;
  const int lh   = lane >> 4;
  const int n0   = wave * 16;
  const int row  = rt * 16 + lm;
  const __bf16* arow = xn + (size_t)row * D_ + lh * 16;
  const __bf16* krow = Wk_b + (size_t)(n0 + lm) * D_ + lh * 16;
  const __bf16* vrow = Wv_b + (size_t)(n0 + lm) * D_ + lh * 16;

  const v8f vzero = {0.f, 0.f, 0.f, 0.f, 0.f, 0.f, 0.f, 0.f};
  v8f ck = vzero, cv = vzero;

  // software pipeline: fetch k0+32 fragments before issuing k0's WMMAs
  v16bf a  = *(const v16bf*)(arow);
  v16bf bk = *(const v16bf*)(krow);
  v16bf bv = *(const v16bf*)(vrow);
  for (int k0 = 0; k0 < D_; k0 += 32) {
    v16bf a2 = a, bk2 = bk, bv2 = bv;
    if (k0 + 32 < D_) {
      a2  = *(const v16bf*)(arow + k0 + 32);
      bk2 = *(const v16bf*)(krow + k0 + 32);
      bv2 = *(const v16bf*)(vrow + k0 + 32);
    }
    ck = __builtin_amdgcn_wmma_f32_16x16x32_bf16(false, a, false, bk, (short)0, ck, false, false);
    cv = __builtin_amdgcn_wmma_f32_16x16x32_bf16(false, a, false, bv, (short)0, cv, false, false);
    a = a2; bk = bk2; bv = bv2;
  }

  const int   sidx = n0 + lm;
  const float ef   = __expf(time_first[sidx]);
#pragma unroll
  for (int r = 0; r < 8; ++r) {
    const int   m   = r + lh * 8;
    const float sig = 1.f / (1.f + __expf(-ck[r]));
    wkv[(size_t)(rt * 16 + m) * S_ + sidx] = __expf(-ef * sig) * cv[r];
  }
}

// ---------------------------------------------------------------------------
// K4: linear recurrence states[t] = states[t-1]*w + wkv[t]; 512 channels.
// ---------------------------------------------------------------------------
__global__ void wkv_scan(const float* __restrict__ wkv,        // [B*T, S]
                         const float* __restrict__ time_decay, // [S]
                         float*       __restrict__ states,     // [B*T, S]
                         float*       __restrict__ last_state) // [B, S]
{
  const int id = blockIdx.x * blockDim.x + threadIdx.x;
  if (id >= B_ * S_) return;
  const int b = id / S_;
  const int s = id - b * S_;
  const float w = __expf(time_decay[s]);

  const float* src = wkv    + (size_t)b * T_ * S_ + s;
  float*       dst = states + (size_t)b * T_ * S_ + s;
  float st = 0.f;
  for (int t = 0; t < T_; ++t) {
    st = st * w + src[(size_t)t * S_];
    dst[(size_t)t * S_] = st;
  }
  last_state[id] = st;
}

// ---------------------------------------------------------------------------
// K5: out = states @ Wo^T  (K = 64 -> two WMMA steps), f32 store.
// ---------------------------------------------------------------------------
__global__ __launch_bounds__(256, 1)
void out_proj(const float*  __restrict__ states, // [B*T, S] f32
              const __bf16* __restrict__ Wo_b,   // [D, S]
              float*        __restrict__ out)    // [B*T, D]
{
  const int rt   = blockIdx.x;
  const int wave = threadIdx.x >> 5;
  const int lane = threadIdx.x & 31;
  const int lm   = lane & 15;
  const int lh   = lane >> 4;
  const int row  = rt * 16 + lm;

  // build A fragments in registers: load f32 states row, convert to bf16
  v16bf a[2];
#pragma unroll
  for (int kk = 0; kk < 2; ++kk) {
    const float* p = states + (size_t)row * S_ + kk * 32 + lh * 16;
#pragma unroll
    for (int i = 0; i < 16; ++i) a[kk][i] = (__bf16)p[i];
  }

  const v8f vzero = {0.f, 0.f, 0.f, 0.f, 0.f, 0.f, 0.f, 0.f};
  v8f acc[16];
#pragma unroll
  for (int j = 0; j < 16; ++j) acc[j] = vzero;

#pragma unroll
  for (int j = 0; j < 16; ++j) {
    const int n0 = (wave + j * 8) * 16;
#pragma unroll
    for (int kk = 0; kk < 2; ++kk) {
      v16bf bm = *(const v16bf*)(Wo_b + (size_t)(n0 + lm) * S_ + kk * 32 + lh * 16);
      acc[j] = __builtin_amdgcn_wmma_f32_16x16x32_bf16(
          false, a[kk], false, bm, (short)0, acc[j], false, false);
    }
  }

#pragma unroll
  for (int j = 0; j < 16; ++j) {
    const int n0 = (wave + j * 8) * 16;
    const int e  = n0 + lm;
#pragma unroll
    for (int r = 0; r < 8; ++r) {
      const int m = r + lh * 8;
      out[(size_t)(rt * 16 + m) * D_ + e] = acc[j][r];
    }
  }
}

// ---------------------------------------------------------------------------
// host launcher
// ---------------------------------------------------------------------------
extern "C" void kernel_launch(void* const* d_in, const int* in_sizes, int n_in,
                              void* d_out, int out_size, void* d_ws, size_t ws_size,
                              hipStream_t stream) {
  (void)in_sizes; (void)n_in; (void)out_size; (void)ws_size;

  // inputs (setup_inputs order)
  const float* x          = (const float*)d_in[0];   // [B,T,D]
  const float* Wk         = (const float*)d_in[1];   // [S,D]
  const float* Wv         = (const float*)d_in[2];   // [S,D]
  const float* Wo         = (const float*)d_in[3];   // [D,S]
  const float* Wts        = (const float*)d_in[4];   // [D,D]
  const float* gate       = (const float*)d_in[5];   // [D]
  const float* time_decay = (const float*)d_in[6];   // [S]
  const float* time_first = (const float*)d_in[7];   // [S]
  const float* ln_w       = (const float*)d_in[8];   // [D]
  const float* ln_b       = (const float*)d_in[9];   // [D]

  float* out        = (float*)d_out;                        // [B,T,D]
  float* last_state = (float*)d_out + (size_t)B_ * T_ * D_; // [B,S]

  // workspace layout (~281 MiB)
  char* ws = (char*)d_ws;
  const size_t N_XD = (size_t)B_ * T_ * D_;   // 67,108,864
  const size_t N_XS = (size_t)B_ * T_ * S_;   //  2,097,152
  __bf16* xb      = (__bf16*)ws;                 ws += N_XD * 2;
  __bf16* xshift  = (__bf16*)ws;                 ws += N_XD * 2;   // -> xn after LN
  __bf16* Wts_b   = (__bf16*)ws;                 ws += (size_t)D_ * D_ * 2;
  __bf16* Wk_b    = (__bf16*)ws;                 ws += (size_t)S_ * D_ * 2;
  __bf16* Wv_b    = (__bf16*)ws;                 ws += (size_t)S_ * D_ * 2;
  __bf16* Wo_b    = (__bf16*)ws;                 ws += (size_t)D_ * S_ * 2;
  float*  wkv     = (float*)ws;                  ws += N_XS * 4;
  float*  states  = (float*)ws;                  /* ws += N_XS * 4; */

  const int CB = 256;
  cvt_f32_bf16<<<(int)((N_XD / 4 + CB - 1) / CB), CB, 0, stream>>>(x, xb, (int)N_XD);
  cvt_f32_bf16<<<(D_ * D_ / 4 + CB - 1) / CB, CB, 0, stream>>>(Wts, Wts_b, D_ * D_);
  cvt_f32_bf16<<<(S_ * D_ / 4 + CB - 1) / CB, CB, 0, stream>>>(Wk, Wk_b, S_ * D_);
  cvt_f32_bf16<<<(S_ * D_ / 4 + CB - 1) / CB, CB, 0, stream>>>(Wv, Wv_b, S_ * D_);
  cvt_f32_bf16<<<(D_ * S_ / 4 + CB - 1) / CB, CB, 0, stream>>>(Wo, Wo_b, D_ * S_);

  // K1: big WMMA GEMM + gate blend (grid: 16 N-tiles x 512 M-tiles)
  dim3 g1(D_ / BN, (B_ * T_) / BM);
  gemm_shift_gate<<<g1, 256, 0, stream>>>(xb, Wts_b, x, gate, xshift);

  // K2: LayerNorm (in place, xshift becomes xn)
  layernorm_inplace<<<B_ * T_, 256, 0, stream>>>(xshift, ln_w, ln_b);

  // K3: k/v WMMA GEMMs + fused wkv elementwise
  const int ROW_TILES = (B_ * T_) / 16;  // 2048
  kv_wkv<<<ROW_TILES, 128, 0, stream>>>(xshift, Wk_b, Wv_b, time_first, wkv);

  // K4: sequential scan over T (512 independent channels)
  wkv_scan<<<(B_ * S_ + CB - 1) / CB, CB, 0, stream>>>(wkv, time_decay, states, last_state);

  // K5: output projection WMMA GEMM
  out_proj<<<ROW_TILES, 256, 0, stream>>>(states, Wo_b, out);
}
